// Self_Attn_13657996001754
// MI455X (gfx1250) — compile-verified, matrix operand order checked by reading, not run on hardware
//
#include <hip/hip_runtime.h>

#define BATCH 4
#define CIN   256
#define NPIX  4096   // 64*64
#define CRED  32

typedef __attribute__((ext_vector_type(16))) _Float16 v16h;
typedef __attribute__((ext_vector_type(8)))  _Float16 v8h;
typedef __attribute__((ext_vector_type(8)))  float    v8f;
typedef __attribute__((ext_vector_type(4)))  unsigned int u32x4;
typedef __attribute__((ext_vector_type(8)))  int      i32x8;
typedef __attribute__((ext_vector_type(4)))  int      i32x4;

// xor-16 cross-lane swap within wave32 via ds_swizzle (xor_mask=0x10, and_mask=0x1f)
__device__ inline float xor16f(float x) {
    return __int_as_float(__builtin_amdgcn_ds_swizzle(__float_as_int(x), 0x401F));
}

// Load a 16-row x 32-col f16 WMMA A/B fragment from a row-major matrix (global or LDS).
// lane l: row = rbase + (l&15); half = l>>4; cols {half*8..+7} and {16+half*8..+7}.
__device__ inline v16h load_frag16x32(const _Float16* base, int rowstride,
                                      int rbase, int colbase, int lane) {
    int r    = rbase + (lane & 15);
    int half = (lane >> 4) & 1;
    const _Float16* p = base + r * rowstride + colbase + half * 8;
    v8h lo = *(const v8h*)(p);
    v8h hi = *(const v8h*)(p + 16);
    return __builtin_shufflevector(lo, hi, 0,1,2,3,4,5,6,7,8,9,10,11,12,13,14,15);
}

// Issue a TDM 2D tile load Global -> LDS (CDNA5 TENSOR_LOAD_TO_LDS, tracked by TENSORcnt).
// Dims/strides in data_size(=2B) units; addresses in bytes. D# built per ISA 8.3/8.4.
__device__ inline void tdm_load_2d(unsigned lds_addr, const void* gptr,
                                   unsigned tensor_d0, unsigned tensor_d1,
                                   unsigned tile_d0, unsigned tile_d1,
                                   unsigned long long stride0) {
    unsigned long long ga = (unsigned long long)(uintptr_t)gptr;
    u32x4 g0;
    g0[0] = 1u;                                    // count=1, user descriptor
    g0[1] = lds_addr;                              // LDS byte address
    g0[2] = (unsigned)(ga & 0xFFFFFFFFu);          // global_addr[31:0]
    g0[3] = (unsigned)((ga >> 32) & 0x01FFFFFFu) | (2u << 30);  // ga[56:32] | type=2
    i32x8 g1;
    g1[0] = (int)(1u << 16);                       // workgroup_mask=0, data_size=1 (2B)
    g1[1] = (int)((tensor_d0 & 0xFFFFu) << 16);    // tensor_dim0[15:0] @ bits 63:48
    g1[2] = (int)((tensor_d0 >> 16) | ((tensor_d1 & 0xFFFFu) << 16));
    g1[3] = (int)((tensor_d1 >> 16) | ((tile_d0 & 0xFFFFu) << 16));
    g1[4] = (int)(tile_d1 & 0xFFFFu);              // tile_dim1; tile_dim2=0 (2D)
    g1[5] = (int)(stride0 & 0xFFFFFFFFull);        // tensor_dim0_stride[31:0]
    g1[6] = (int)((stride0 >> 32) & 0xFFFFull);    // stride[47:32]; dim1_stride=0
    g1[7] = 0;
    i32x4 z = {0, 0, 0, 0};                        // groups 2/3: zero (tile dims 3/4 unused)
    asm volatile("tensor_load_to_lds %0, %1, %2, %3"
                 :: "s"(g0), "s"(g1), "s"(z), "s"(z) : "memory");
}

// ---------------- Kernel 1: fused QKV 1x1-conv projection ----------------
// q/k stored [b][n][32] f16 (row-major per pixel); v stored transposed [b][32][n] f16.
__global__ __launch_bounds__(256)
void qkv_proj_kernel(const float* __restrict__ x,
                     const float* __restrict__ Wq, const float* __restrict__ bq,
                     const float* __restrict__ Wk, const float* __restrict__ bk,
                     const float* __restrict__ Wv, const float* __restrict__ bv,
                     _Float16* __restrict__ Qh, _Float16* __restrict__ Kh,
                     _Float16* __restrict__ Vt) {
    int t = blockIdx.x * 256 + threadIdx.x;       // t = ((b*32 + o)*4096) + n
    int n = t & (NPIX - 1);
    int o = (t >> 12) & (CRED - 1);
    int b = t >> 17;
    const float* xb = x + (size_t)b * CIN * NPIX + n;
    const float* wq = Wq + o * CIN;
    const float* wk = Wk + o * CIN;
    const float* wv = Wv + o * CIN;
    float aq = 0.f, ak = 0.f, av = 0.f;
    #pragma unroll 8
    for (int c = 0; c < CIN; ++c) {
        float xv = xb[(size_t)c * NPIX];
        aq = fmaf(wq[c], xv, aq);
        ak = fmaf(wk[c], xv, ak);
        av = fmaf(wv[c], xv, av);
    }
    Qh[((size_t)b * NPIX + n) * CRED + o] = (_Float16)(aq + bq[o]);
    Kh[((size_t)b * NPIX + n) * CRED + o] = (_Float16)(ak + bk[o]);
    Vt[((size_t)b * CRED + o) * NPIX + n] = (_Float16)(av + bv[o]);
}

// ---------------- Kernel 2: flash attention ----------------
// Block = 8 waves = 128 queries. K/V tiles (64 keys) staged in LDS by TDM,
// double-buffered, issued by wave 0; all 8 waves consume from LDS.
__global__ __launch_bounds__(256)
void flash_attn_kernel(const _Float16* __restrict__ Qh,
                       const _Float16* __restrict__ Kh,
                       const _Float16* __restrict__ Vt,
                       float* __restrict__ AOut) {   // AOut [b][32][n] f32
    __shared__ alignas(16) _Float16 Ks[2][64 * 32];  // [key][dim]
    __shared__ alignas(16) _Float16 Vs[2][32 * 64];  // [dim][key]

    int lane  = threadIdx.x & 31;
    int wave  = threadIdx.x >> 5;
    int b     = blockIdx.y;
    int qbase = blockIdx.x * 128 + wave * 16;

    const _Float16* Qb = Qh + (size_t)b * NPIX * CRED;
    const _Float16* Kb = Kh + (size_t)b * NPIX * CRED;
    const _Float16* Vb = Vt + (size_t)b * CRED * NPIX;

    // B operand: Q^T (32 dims x 16 queries) — loop invariant, from global
    v16h bq = load_frag16x32(Qb, CRED, qbase, 0, lane);

    v8f acc0 = {};  // O^T dims 0..15  (lane = query, vgpr = dim)
    v8f acc1 = {};  // O^T dims 16..31
    float m = -3.0e38f, lsum = 0.0f;

    // Prologue: stage tile 0
    if (wave == 0) {
        tdm_load_2d((unsigned)(uintptr_t)&Ks[0][0], Kb, CRED, NPIX, CRED, 64, CRED);
        tdm_load_2d((unsigned)(uintptr_t)&Vs[0][0], Vb, NPIX, CRED, 64, CRED, NPIX);
    }

    int cur = 0;
    for (int kb = 0; kb < NPIX; kb += 64) {
        if (wave == 0) {
            if (kb + 64 < NPIX) {   // prefetch next stage into the other buffer
                int nxt = cur ^ 1;
                tdm_load_2d((unsigned)(uintptr_t)&Ks[nxt][0], Kb + (size_t)(kb + 64) * CRED,
                            CRED, NPIX, CRED, 64, CRED);
                tdm_load_2d((unsigned)(uintptr_t)&Vs[nxt][0], Vb + (kb + 64),
                            NPIX, CRED, 64, CRED, NPIX);
                __builtin_amdgcn_s_wait_tensorcnt(2);  // current stage's 2 loads done
            } else {
                __builtin_amdgcn_s_wait_tensorcnt(0);
            }
        }
        __syncthreads();   // LDS tile [cur] ready for all waves

        const _Float16* kt = &Ks[cur][0];
        const _Float16* vt = &Vs[cur][0];

        // S^T = K_tile x Q^T : four 16(key)x16(query) fragments covering 64 keys
        v16h ak0 = load_frag16x32(kt, CRED, 0,  0, lane);
        v16h ak1 = load_frag16x32(kt, CRED, 16, 0, lane);
        v16h ak2 = load_frag16x32(kt, CRED, 32, 0, lane);
        v16h ak3 = load_frag16x32(kt, CRED, 48, 0, lane);
        v8f z = {};
        v8f s0 = __builtin_amdgcn_wmma_f32_16x16x32_f16(false, ak0, false, bq, (short)0, z, false, false);
        v8f s1 = __builtin_amdgcn_wmma_f32_16x16x32_f16(false, ak1, false, bq, (short)0, z, false, false);
        v8f s2 = __builtin_amdgcn_wmma_f32_16x16x32_f16(false, ak2, false, bq, (short)0, z, false, false);
        v8f s3 = __builtin_amdgcn_wmma_f32_16x16x32_f16(false, ak3, false, bq, (short)0, z, false, false);

        // online softmax over 64 keys: per-lane partial (32 keys), xor16 for the rest
        float cm = s0[0];
        #pragma unroll
        for (int i = 1; i < 8; ++i) cm = fmaxf(cm, s0[i]);
        #pragma unroll
        for (int i = 0; i < 8; ++i) cm = fmaxf(cm, s1[i]);
        #pragma unroll
        for (int i = 0; i < 8; ++i) cm = fmaxf(cm, s2[i]);
        #pragma unroll
        for (int i = 0; i < 8; ++i) cm = fmaxf(cm, s3[i]);
        cm = fmaxf(cm, xor16f(cm));
        float mnew  = fmaxf(m, cm);
        float scale = __expf(m - mnew);

        float rs = 0.0f;
        v16h bp0, bp1;  // P^T as B operands (2 x 32keys x 16queries), packed in-lane
        #pragma unroll
        for (int i = 0; i < 8; ++i) {
            float p0 = __expf(s0[i] - mnew);
            float p1 = __expf(s1[i] - mnew);
            float p2 = __expf(s2[i] - mnew);
            float p3 = __expf(s3[i] - mnew);
            rs += (p0 + p1) + (p2 + p3);
            bp0[i] = (_Float16)p0;  bp0[i + 8] = (_Float16)p1;
            bp1[i] = (_Float16)p2;  bp1[i + 8] = (_Float16)p3;
        }
        rs  += xor16f(rs);
        lsum = lsum * scale + rs;
        m    = mnew;

        #pragma unroll
        for (int i = 0; i < 8; ++i) { acc0[i] *= scale; acc1[i] *= scale; }

        // O^T += V^T_tile x P^T   (Vs is [dim][key], rowstride 64)
        v16h av00 = load_frag16x32(vt, 64, 0,  0,  lane);
        v16h av10 = load_frag16x32(vt, 64, 16, 0,  lane);
        v16h av01 = load_frag16x32(vt, 64, 0,  32, lane);
        v16h av11 = load_frag16x32(vt, 64, 16, 32, lane);
        acc0 = __builtin_amdgcn_wmma_f32_16x16x32_f16(false, av00, false, bp0, (short)0, acc0, false, false);
        acc1 = __builtin_amdgcn_wmma_f32_16x16x32_f16(false, av10, false, bp0, (short)0, acc1, false, false);
        acc0 = __builtin_amdgcn_wmma_f32_16x16x32_f16(false, av01, false, bp1, (short)0, acc0, false, false);
        acc1 = __builtin_amdgcn_wmma_f32_16x16x32_f16(false, av11, false, bp1, (short)0, acc1, false, false);

        __syncthreads();   // all waves done with tile [cur] before it is overwritten
        cur ^= 1;
    }

    float inv  = 1.0f / lsum;
    int   q    = qbase + (lane & 15);
    int   half = (lane >> 4) & 1;
    float* Ab  = AOut + (size_t)b * CRED * NPIX;
    #pragma unroll
    for (int r = 0; r < 8; ++r) {
        int d = r + 8 * half;
        Ab[(size_t)d        * NPIX + q] = acc0[r] * inv;
        Ab[(size_t)(16 + d) * NPIX + q] = acc1[r] * inv;
    }
}

// ---------------- Kernel 3: output 1x1-conv projection ----------------
__global__ __launch_bounds__(256)
void out_proj_kernel(const float* __restrict__ AOut, const float* __restrict__ Wo,
                     const float* __restrict__ bo, float* __restrict__ out) {
    int t = blockIdx.x * 256 + threadIdx.x;       // t = ((b*256 + c)*4096) + n
    int n = t & (NPIX - 1);
    int c = (t >> 12) & (CIN - 1);
    int b = t >> 20;
    const float* Ab = AOut + (size_t)b * CRED * NPIX + n;
    const float* wo = Wo + c * CRED;
    float acc = bo[c];
    #pragma unroll
    for (int o = 0; o < CRED; ++o)
        acc = fmaf(wo[o], Ab[(size_t)o * NPIX], acc);
    out[((size_t)b * CIN + c) * NPIX + n] = acc;
}

extern "C" void kernel_launch(void* const* d_in, const int* in_sizes, int n_in,
                              void* d_out, int out_size, void* d_ws, size_t ws_size,
                              hipStream_t stream) {
    const float* x  = (const float*)d_in[0];
    const float* Wq = (const float*)d_in[1];
    const float* bq = (const float*)d_in[2];
    const float* Wk = (const float*)d_in[3];
    const float* bk = (const float*)d_in[4];
    const float* Wv = (const float*)d_in[5];
    const float* bv = (const float*)d_in[6];
    const float* Wo = (const float*)d_in[7];
    const float* bo = (const float*)d_in[8];
    float* out = (float*)d_out;

    const size_t qkvElems = (size_t)BATCH * NPIX * CRED;   // 524288
    _Float16* Qh = (_Float16*)d_ws;
    _Float16* Kh = Qh + qkvElems;
    _Float16* Vt = Kh + qkvElems;
    float*  AOut = (float*)(Vt + qkvElems);                // 2 MB f32, total ws = 5 MB

    // 1) QKV projection: B*CR*N threads
    qkv_proj_kernel<<<(BATCH * CRED * NPIX) / 256, 256, 0, stream>>>(
        x, Wq, bq, Wk, bk, Wv, bv, Qh, Kh, Vt);

    // 2) Flash attention: grid (N/128 qblocks, B), 8 waves/block, 16 queries/wave
    flash_attn_kernel<<<dim3(NPIX / 128, BATCH), 256, 0, stream>>>(Qh, Kh, Vt, AOut);

    // 3) Output projection: B*C*N threads
    out_proj_kernel<<<(BATCH * CIN * NPIX) / 256, 256, 0, stream>>>(AOut, Wo, bo, out);
}
